// MultiHeadAttentionV2_24120536334834
// MI455X (gfx1250) — matrix operand order repaired and checked by reading, not compiled, not run on hardware
//
#include <hip/hip_runtime.h>

// ---------------------------------------------------------------------------
// Types for CDNA5 WMMA (wave32): bf16 A/B fragments, f32 accumulators.
// ---------------------------------------------------------------------------
typedef __bf16 v16bf __attribute__((ext_vector_type(16)));
typedef float  v8f   __attribute__((ext_vector_type(8)));
typedef unsigned int u32x4 __attribute__((ext_vector_type(4)));
typedef int          i32x4 __attribute__((ext_vector_type(4)));
typedef int          i32x8 __attribute__((ext_vector_type(8)));

union Frag {            // 16 bf16 = 32 bytes = two uint4 LDS loads
  uint4 u[2];
  v16bf v;
};
union Acc {             // 16x16 f32 C/D tile slice per lane
  v8f   v;
  float f[8];
};

__device__ __forceinline__ unsigned short f2bf(float x) {
  unsigned int u = __float_as_uint(x);
  unsigned int r = u + 0x7FFFu + ((u >> 16) & 1u);   // round-to-nearest-even
  return (unsigned short)(r >> 16);
}

// ---------------------------------------------------------------------------
// Tensor Data Mover: 2D tile (bf16) from global -> LDS.  D# per ISA ch.8.
// Issued by wave 0; completion via s_wait_tensorcnt + workgroup barrier.
// Double-buffered: next tile's DMA is issued before computing current tile.
// ---------------------------------------------------------------------------
#if __has_builtin(__builtin_amdgcn_tensor_load_to_lds)
#define HAVE_TDM 1
#else
#define HAVE_TDM 0
#warning "CDNA5 probe: __builtin_amdgcn_tensor_load_to_lds NOT available; using plain-load fallback"
#endif

#if __has_builtin(__builtin_amdgcn_s_wait_tensorcnt)
#define WAIT_TENSOR() __builtin_amdgcn_s_wait_tensorcnt(0)
#else
#define WAIT_TENSOR() asm volatile("s_wait_tensorcnt 0x0" ::: "memory")
#warning "CDNA5 probe: __builtin_amdgcn_s_wait_tensorcnt NOT available; using inline asm"
#endif

#if HAVE_TDM
__device__ __forceinline__ void tdm_load_2d(unsigned lds_addr, const void* gptr,
                                            unsigned tensor_d0, unsigned tensor_d1,
                                            unsigned tile_d0, unsigned tile_d1,
                                            unsigned stride0) {
  unsigned long long ga = (unsigned long long)(uintptr_t)gptr;
  u32x4 g0;
  g0.x = 1u;                                            // count=1, user mode
  g0.y = lds_addr;                                      // LDS byte address
  g0.z = (unsigned)(ga & 0xFFFFFFFFu);                  // global_addr[31:0]
  g0.w = (unsigned)((ga >> 32) & 0x01FFFFFFu) | (2u << 30);  // [56:32] | type=2
  i32x8 g1;
  g1[0] = (int)(1u << 16);                              // data_size=1 -> 2 bytes
  g1[1] = (int)((tensor_d0 & 0xFFFFu) << 16);           // tensor_dim0 lo
  g1[2] = (int)((tensor_d0 >> 16) | ((tensor_d1 & 0xFFFFu) << 16));
  g1[3] = (int)((tensor_d1 >> 16) | (tile_d0 << 16));   // tile_dim0
  g1[4] = (int)(tile_d1 & 0xFFFFu);                     // tile_dim1, tile_dim2=0
  g1[5] = (int)stride0;                                 // tensor_dim0_stride lo
  g1[6] = 0;
  g1[7] = 0;
  i32x4 z4 = {0, 0, 0, 0};
#if defined(__clang_major__) && __clang_major__ >= 23
  i32x8 z8 = {0, 0, 0, 0, 0, 0, 0, 0};
  __builtin_amdgcn_tensor_load_to_lds(g0, g1, z4, z4, z8, 0);
#else
  __builtin_amdgcn_tensor_load_to_lds(g0, g1, z4, z4, 0);
#endif
}
#endif

__device__ __forceinline__ unsigned lds_off(const void* p) {
  return (unsigned)(uintptr_t)p;    // LDS aperture: addr[31:0] = LDS offset
}

// ---------------------------------------------------------------------------
// Kernel 1a: fp32 -> bf16 bulk convert (vectorized), for x
// ---------------------------------------------------------------------------
__global__ void f32_to_bf16(const float* __restrict__ src,
                            unsigned short* __restrict__ dst, int n4) {
  int i = blockIdx.x * blockDim.x + threadIdx.x;
  if (i < n4) {
    float4 f = ((const float4*)src)[i];
    ushort4 o;
    o.x = f2bf(f.x); o.y = f2bf(f.y); o.z = f2bf(f.z); o.w = f2bf(f.w);
    ((ushort4*)dst)[i] = o;
  }
}

// ---------------------------------------------------------------------------
// Kernel 1b: fp32 [K][N] -> bf16 transposed [N][K] (one-time weight prep, so
// all GEMM B-tiles become contiguous row-major reads; no LDS scatter).
// ---------------------------------------------------------------------------
__global__ void f32_to_bf16_T(const float* __restrict__ src,
                              unsigned short* __restrict__ dst, int K, int N) {
  int idx = blockIdx.x * blockDim.x + threadIdx.x;
  if (idx < K * N) {
    int k = idx / N, n = idx - k * N;                   // coalesced read
    dst[(size_t)n * K + k] = f2bf(src[idx]);
  }
}

// ---------------------------------------------------------------------------
// One GEMM K-step: fragment loads from LDS + 8 WMMA (64x32 per wave).
// A frags: lane row=M, chunks {K0-7,K16-23}/{K8-15,K24-31} per ISA 7.12.2.
// ---------------------------------------------------------------------------
__device__ __forceinline__ void gemm_step(Acc (&acc)[4][2],
                                          const unsigned short* __restrict__ AsT,
                                          const unsigned short* __restrict__ BsT,
                                          int wm, int wn, int l15, int half) {
  Frag af[4], bfr[2];
#pragma unroll
  for (int i = 0; i < 4; ++i) {
    const unsigned short* p = &AsT[(wm * 64 + i * 16 + l15) * 32 + (half ? 8 : 0)];
    af[i].u[0] = *(const uint4*)(p);
    af[i].u[1] = *(const uint4*)(p + 16);
  }
#pragma unroll
  for (int j = 0; j < 2; ++j) {
    const unsigned short* p = &BsT[(wn * 32 + j * 16 + l15) * 32 + (half ? 8 : 0)];
    bfr[j].u[0] = *(const uint4*)(p);
    bfr[j].u[1] = *(const uint4*)(p + 16);
  }
#pragma unroll
  for (int i = 0; i < 4; ++i)
#pragma unroll
    for (int j = 0; j < 2; ++j)
      acc[i][j].v = __builtin_amdgcn_wmma_f32_16x16x32_bf16(
          false, af[i].v, false, bfr[j].v, (short)0, acc[i][j].v, false, false);
}

// ---------------------------------------------------------------------------
// Kernel 2/4: bf16 WMMA GEMM.  C = A[MxK] * Bt[NxK]^T.  128x128 tile,
// 256 threads (8 waves), 64x32/wave, K-step 32, double-buffered TDM staging.
// MODE 0: bf16 -> [B,H,S,HD]                      (K projection)
// MODE 3: bf16 -> [B,H,S,HD], *(1/8)*log2(e)      (Q projection, pre-scaled)
// MODE 2: bf16 -> [B,H,HD,S]                      (V projection, transposed)
// MODE 1: fp32 + bias, row-major                  (output projection)
// ---------------------------------------------------------------------------
template <int MODE>
__global__ __launch_bounds__(256, 1)
void gemm_bf16_128(const unsigned short* __restrict__ A,
                   const unsigned short* __restrict__ Bt,
                   unsigned short* __restrict__ Dbf,
                   float* __restrict__ Dfp,
                   const float* __restrict__ bias,
                   int M, int N, int K) {
  __shared__ unsigned short As[2][128 * 32];   // [m][k], 8 KB each
  __shared__ unsigned short Bs[2][128 * 32];   // [n][k], 8 KB each

  const int t    = threadIdx.x;
  const int lane = t & 31;
  const int wave = t >> 5;      // 0..7
  const int wm   = wave >> 2;   // 0..1
  const int wn   = wave & 3;    // 0..3
  const int l15  = lane & 15;
  const int half = lane >> 4;
  const int tm   = blockIdx.x * 128;
  const int tn   = blockIdx.y * 128;

  Acc acc[4][2];
#pragma unroll
  for (int i = 0; i < 4; ++i)
#pragma unroll
    for (int j = 0; j < 2; ++j)
#pragma unroll
      for (int r = 0; r < 8; ++r) acc[i][j].f[r] = 0.0f;

#if HAVE_TDM
  if (wave == 0) {                               // prologue: stage tile 0
    tdm_load_2d(lds_off(As[0]), A  + (size_t)tm * K,
                (unsigned)K, (unsigned)M, 32u, 128u, (unsigned)K);
    tdm_load_2d(lds_off(Bs[0]), Bt + (size_t)tn * K,
                (unsigned)K, (unsigned)N, 32u, 128u, (unsigned)K);
    WAIT_TENSOR();
  }
  __syncthreads();
  int cur = 0;
  for (int kk = 0; kk < K; kk += 32) {
    if (wave == 0 && kk + 32 < K) {              // async-prefetch next tile
      tdm_load_2d(lds_off(As[cur ^ 1]), A  + (size_t)tm * K + kk + 32,
                  (unsigned)K, (unsigned)M, 32u, 128u, (unsigned)K);
      tdm_load_2d(lds_off(Bs[cur ^ 1]), Bt + (size_t)tn * K + kk + 32,
                  (unsigned)K, (unsigned)N, 32u, 128u, (unsigned)K);
    }
    gemm_step(acc, As[cur], Bs[cur], wm, wn, l15, half);
    __syncthreads();                             // all done reading cur
    if (wave == 0) WAIT_TENSOR();                // next tile landed
    __syncthreads();                             // publish
    cur ^= 1;
  }
#else
  for (int kk = 0; kk < K; kk += 32) {
    __syncthreads();
#pragma unroll
    for (int c = 0; c < 2; ++c) {
      int chunk = t * 2 + c;                     // 512 chunks of 8 bf16
      int row   = chunk >> 2;
      int col   = (chunk & 3) * 8;
      *(uint4*)(&As[0][row * 32 + col]) =
          *(const uint4*)(A + (size_t)(tm + row) * K + kk + col);
      *(uint4*)(&Bs[0][row * 32 + col]) =
          *(const uint4*)(Bt + (size_t)(tn + row) * K + kk + col);
    }
    __syncthreads();
    gemm_step(acc, As[0], Bs[0], wm, wn, l15, half);
  }
#endif

  // C layout: VGPR r, lanes 0-15 -> (M=r, N=lane); lanes 16-31 -> (M=r+8)
#pragma unroll
  for (int i = 0; i < 4; ++i)
#pragma unroll
    for (int j = 0; j < 2; ++j)
#pragma unroll
      for (int r = 0; r < 8; ++r) {
        int row = tm + wm * 64 + i * 16 + (half ? r + 8 : r);
        int col = tn + wn * 32 + j * 16 + l15;
        float val = acc[i][j].f[r];
        if (MODE == 0 || MODE == 3) {        // [b][h][s][hd]
          if (MODE == 3) val *= 0.18033688011112042f;   // (1/8)*log2(e)
          int b = row >> 11, s = row & 2047;
          int h = col >> 6,  hd = col & 63;
          Dbf[(((size_t)b * 16 + h) * 2048 + s) * 64 + hd] = f2bf(val);
        } else if (MODE == 2) {              // [b][h][hd][s]  (V transposed)
          int b = row >> 11, s = row & 2047;
          int h = col >> 6,  hd = col & 63;
          Dbf[(((size_t)b * 16 + h) * 64 + hd) * 2048 + s] = f2bf(val);
        } else {
          Dfp[(size_t)row * N + col] = val + bias[col];
        }
      }
}

// ---------------------------------------------------------------------------
// Flash-attention inner tile: QK^T (WMMA) -> base-2 online softmax -> PV.
// Q pre-scaled by (1/8)*log2(e), so softmax uses exp2f directly.
// ---------------------------------------------------------------------------
struct AttnState {
  Frag  qf[2];
  Acc   cacc[4];
  float mrow[8], lrow[8];
};

__device__ __forceinline__ void attn_tile(AttnState& st,
                                          const unsigned short* __restrict__ KsT,
                                          const unsigned short* __restrict__ VtsT,
                                          unsigned short* __restrict__ PsW,
                                          int kv, int qbase, int l15, int half) {
  // scores: two 16-wide N-tiles, two K=32 steps over d
  Acc sc[2];
#pragma unroll
  for (int n2 = 0; n2 < 2; ++n2) {
#pragma unroll
    for (int r = 0; r < 8; ++r) sc[n2].f[r] = 0.0f;
#pragma unroll
    for (int dc = 0; dc < 2; ++dc) {
      Frag kf;
      const unsigned short* p = &KsT[(n2 * 16 + l15) * 64 + dc * 32 + (half ? 8 : 0)];
      kf.u[0] = *(const uint4*)(p);
      kf.u[1] = *(const uint4*)(p + 16);
      sc[n2].v = __builtin_amdgcn_wmma_f32_16x16x32_bf16(
          false, st.qf[dc].v, false, kf.v, (short)0, sc[n2].v, false, false);
    }
  }

  // causal mask + base-2 online softmax (each row lives across 16 lanes)
#pragma unroll
  for (int r = 0; r < 8; ++r) {
    int   qg = qbase + (half ? r + 8 : r);
    float s0 = sc[0].f[r];
    float s1 = sc[1].f[r];
    if (kv + l15      > qg) s0 = -1e30f;
    if (kv + 16 + l15 > qg) s1 = -1e30f;
    float tmax = fmaxf(s0, s1);
#pragma unroll
    for (int m = 1; m < 16; m <<= 1)
      tmax = fmaxf(tmax, __shfl_xor(tmax, m, 32));
    float mnew  = fmaxf(st.mrow[r], tmax);
    float alpha = exp2f(st.mrow[r] - mnew);
    float p0    = exp2f(s0 - mnew);
    float p1    = exp2f(s1 - mnew);
    float psum  = p0 + p1;
#pragma unroll
    for (int m = 1; m < 16; m <<= 1)
      psum += __shfl_xor(psum, m, 32);
    st.lrow[r] = st.lrow[r] * alpha + psum;
    st.mrow[r] = mnew;
#pragma unroll
    for (int j = 0; j < 4; ++j) st.cacc[j].f[r] *= alpha;
    // stage P (C layout) to LDS as bf16 for A-layout reload
    int prow = half ? r + 8 : r;
    PsW[prow * 32 + l15]      = f2bf(p0);
    PsW[prow * 32 + 16 + l15] = f2bf(p1);
  }

  // reload P as A-fragment (same-wave LDS ops are in-order)
  Frag pf;
  {
    const unsigned short* p = &PsW[l15 * 32 + (half ? 8 : 0)];
    pf.u[0] = *(const uint4*)(p);
    pf.u[1] = *(const uint4*)(p + 16);
  }
  // ctx += P @ V  (B-frags = kj {0-7,16-23}/{8-15,24-31} from transposed V tile)
#pragma unroll
  for (int j = 0; j < 4; ++j) {
    Frag vf;
    const unsigned short* p = &VtsT[(j * 16 + l15) * 32 + (half ? 8 : 0)];
    vf.u[0] = *(const uint4*)(p);
    vf.u[1] = *(const uint4*)(p + 16);
    st.cacc[j].v = __builtin_amdgcn_wmma_f32_16x16x32_bf16(
        false, pf.v, false, vf.v, (short)0, st.cacc[j].v, false, false);
  }
}

// ---------------------------------------------------------------------------
// Kernel 3: causal flash attention.  One (b,h), one 64-row q-block per WG.
// 128 threads = 4 waves, 16 q-rows/wave.  KV tiles of 32, double-buffered TDM.
// ---------------------------------------------------------------------------
__global__ __launch_bounds__(128, 1)
void flash_attn(const unsigned short* __restrict__ Q,
                const unsigned short* __restrict__ Kb,
                const unsigned short* __restrict__ V,
                unsigned short* __restrict__ ctx) {
  __shared__ unsigned short Qs[64 * 64];          // [qrow][d]
  __shared__ unsigned short Ks[2][32 * 64];       // [kj][d]
  __shared__ unsigned short Vts[2][64 * 32];      // [d][kj]
  __shared__ unsigned short Ps[4 * 16 * 32];      // per-wave P staging

  const int t    = threadIdx.x;
  const int lane = t & 31;
  const int wave = t >> 5;      // 0..3
  const int l15  = lane & 15;
  const int half = lane >> 4;
  const int bh   = blockIdx.y;
  const int b    = bh >> 4;     // H = 16
  const int h    = bh & 15;
  const int qb   = blockIdx.x;

  const size_t base = (size_t)bh * 2048 * 64;
  const unsigned short* Qp = Q  + base;           // [2048][64]
  const unsigned short* Kp = Kb + base;           // [2048][64]
  const unsigned short* Vp = V  + base;           // [64][2048]
  unsigned short* PsW = &Ps[wave * 512];

  const int kend = (qb + 1) * 64;

#if HAVE_TDM
  if (wave == 0) {                                // prologue: Q + first K/V
    tdm_load_2d(lds_off(Qs), Qp + (size_t)qb * 64 * 64, 64u, 2048u, 64u, 64u, 64u);
    tdm_load_2d(lds_off(Ks[0]),  Kp, 64u, 2048u, 64u, 32u, 64u);
    tdm_load_2d(lds_off(Vts[0]), Vp, 2048u, 64u, 32u, 64u, 2048u);
    WAIT_TENSOR();
  }
  __syncthreads();
#else
#pragma unroll
  for (int c = 0; c < 4; ++c) {
    int chunk = t * 4 + c;
    int row   = chunk >> 3;
    int col   = (chunk & 7) * 8;
    *(uint4*)(&Qs[row * 64 + col]) =
        *(const uint4*)(Qp + (size_t)(qb * 64 + row) * 64 + col);
  }
#pragma unroll
  for (int c = 0; c < 2; ++c) {
    int chunk = t * 2 + c;
    int row   = chunk >> 3;
    int col   = (chunk & 7) * 8;
    *(uint4*)(&Ks[0][row * 64 + col]) = *(const uint4*)(Kp + (size_t)row * 64 + col);
  }
#pragma unroll
  for (int c = 0; c < 2; ++c) {
    int chunk = t * 2 + c;
    int d     = chunk >> 2;
    int col   = (chunk & 3) * 8;
    *(uint4*)(&Vts[0][d * 32 + col]) = *(const uint4*)(Vp + (size_t)d * 2048 + col);
  }
  __syncthreads();
#endif

  AttnState st;
  {
    int row = wave * 16 + l15;                    // Q frags live all loop
#pragma unroll
    for (int dc = 0; dc < 2; ++dc) {
      const unsigned short* p = &Qs[row * 64 + dc * 32 + (half ? 8 : 0)];
      st.qf[dc].u[0] = *(const uint4*)(p);
      st.qf[dc].u[1] = *(const uint4*)(p + 16);
    }
  }
#pragma unroll
  for (int j = 0; j < 4; ++j)
#pragma unroll
    for (int r = 0; r < 8; ++r) st.cacc[j].f[r] = 0.0f;
#pragma unroll
  for (int r = 0; r < 8; ++r) { st.mrow[r] = -1e30f; st.lrow[r] = 0.0f; }

  const int qbase = qb * 64 + wave * 16;

#if HAVE_TDM
  int cur = 0;
  for (int kv = 0; kv < kend; kv += 32) {
    if (wave == 0 && kv + 32 < kend) {            // async-prefetch next K/V
      tdm_load_2d(lds_off(Ks[cur ^ 1]),  Kp + (size_t)(kv + 32) * 64,
                  64u, 2048u, 64u, 32u, 64u);
      tdm_load_2d(lds_off(Vts[cur ^ 1]), Vp + (size_t)(kv + 32),
                  2048u, 64u, 32u, 64u, 2048u);
    }
    attn_tile(st, Ks[cur], Vts[cur], PsW, kv, qbase, l15, half);
    __syncthreads();                              // all done reading cur
    if (wave == 0) WAIT_TENSOR();                 // next tile landed
    __syncthreads();                              // publish
    cur ^= 1;
  }
#else
  for (int kv = 0; kv < kend; kv += 32) {
    if (kv > 0) {
      __syncthreads();
#pragma unroll
      for (int c = 0; c < 2; ++c) {
        int chunk = t * 2 + c;
        int row   = chunk >> 3;
        int col   = (chunk & 7) * 8;
        *(uint4*)(&Ks[0][row * 64 + col]) =
            *(const uint4*)(Kp + (size_t)(kv + row) * 64 + col);
      }
#pragma unroll
      for (int c = 0; c < 2; ++c) {
        int chunk = t * 2 + c;
        int d     = chunk >> 2;
        int col   = (chunk & 3) * 8;
        *(uint4*)(&Vts[0][d * 32 + col]) =
            *(const uint4*)(Vp + (size_t)d * 2048 + kv + col);
      }
      __syncthreads();
    }
    attn_tile(st, Ks[0], Vts[0], PsW, kv, qbase, l15, half);
  }
#endif

  // finalize: divide by row sum, store ctx as bf16 in [b, s, h*64+d]
#pragma unroll
  for (int r = 0; r < 8; ++r) {
    float  inv = 1.0f / st.lrow[r];
    int    qg  = qbase + (half ? r + 8 : r);
    size_t rb  = ((size_t)b * 2048 + qg) * 1024 + (size_t)h * 64;
#pragma unroll
    for (int j = 0; j < 4; ++j)
      ctx[rb + j * 16 + l15] = f2bf(st.cacc[j].f[r] * inv);
  }
}

// ---------------------------------------------------------------------------
// Launch: convert (+weight transpose) -> QKV GEMMs -> flash attn -> out GEMM
// ---------------------------------------------------------------------------
extern "C" void kernel_launch(void* const* d_in, const int* in_sizes, int n_in,
                              void* d_out, int out_size, void* d_ws, size_t ws_size,
                              hipStream_t stream) {
  (void)in_sizes; (void)n_in; (void)out_size; (void)ws_size;
  const float* x  = (const float*)d_in[0];
  const float* Wq = (const float*)d_in[1];
  const float* Wk = (const float*)d_in[2];
  const float* Wv = (const float*)d_in[3];
  const float* Wo = (const float*)d_in[4];
  const float* bo = (const float*)d_in[5];
  float* out = (float*)d_out;

  const int S = 2048, D = 1024;
  const size_t MX = (size_t)4 * S;        // 8192 rows
  const size_t DD = (size_t)D * D;        // 1M

  unsigned short* ws   = (unsigned short*)d_ws;
  unsigned short* xb   = ws;              // 8M bf16 [M][K]
  unsigned short* wqt  = xb  + MX * D;    // [N][K] transposed weights
  unsigned short* wkt  = wqt + DD;
  unsigned short* wvt  = wkt + DD;
  unsigned short* wot  = wvt + DD;
  unsigned short* Qb   = wot + DD;        // [B,H,S,HD], pre-scaled
  unsigned short* Kbf  = Qb  + MX * D;    // [B,H,S,HD]
  unsigned short* Vb   = Kbf + MX * D;    // [B,H,HD,S]
  unsigned short* ctxb = Vb  + MX * D;    // [B,S,D]

  {
    int n4 = (int)(MX * D / 4);
    f32_to_bf16<<<(n4 + 255) / 256, 256, 0, stream>>>(x, xb, n4);
  }
  int nW = (int)DD;
  f32_to_bf16_T<<<(nW + 255) / 256, 256, 0, stream>>>(Wq, wqt, D, D);
  f32_to_bf16_T<<<(nW + 255) / 256, 256, 0, stream>>>(Wk, wkt, D, D);
  f32_to_bf16_T<<<(nW + 255) / 256, 256, 0, stream>>>(Wv, wvt, D, D);
  f32_to_bf16_T<<<(nW + 255) / 256, 256, 0, stream>>>(Wo, wot, D, D);

  dim3 g((unsigned)(MX / 128), (unsigned)(D / 128));
  gemm_bf16_128<3><<<g, 256, 0, stream>>>(xb, wqt, Qb,  nullptr, nullptr, (int)MX, D, D);
  gemm_bf16_128<0><<<g, 256, 0, stream>>>(xb, wkt, Kbf, nullptr, nullptr, (int)MX, D, D);
  gemm_bf16_128<2><<<g, 256, 0, stream>>>(xb, wvt, Vb,  nullptr, nullptr, (int)MX, D, D);

  dim3 ga(S / 64, 4 * 16);
  flash_attn<<<ga, 128, 0, stream>>>(Qb, Kbf, Vb, ctxb);

  gemm_bf16_128<1><<<g, 256, 0, stream>>>(ctxb, wot, nullptr, out, bo, (int)MX, D, D);
}
